// GraphIpaFrameDenoiser_31112743092521
// MI455X (gfx1250) — matrix operand order repaired and checked by reading, not compiled, not run on hardware
//
// CDNA5 (gfx1250) implementation of GraphIpaFrameDenoiser forward.
//
// All dense linears (~140 GFLOP, dominated by the edge-transition MLPs over
// 131072 edges) run through a tiled v_wmma_f32_16x16x32_bf16 GEMM:
//  - weights pre-packed to bf16 in WMMA B-fragment order (2 x b128 loads/lane)
//  - activations loaded as contiguous float4 pairs, converted inline to bf16
//  - 4 M-tiles per wave (64x16 C per wave), branch-free K loop (K % 32 == 0;
//    the lone K=321 GEMM is zero-padded to 352)
// Irregular work (segment softmax over random edges, quaternion frame
// updates, RBF/sinusoid features) is scalar + global atomics.
//
// Input pointer order assumption: setup_inputs() dict order, 'params' pytree
// flattened depth-first in insertion order (see parse code below).
//
// Workspace: ~420 MB peak (two E_SP x 256 f32 edge buffers ping-pong; the
// IPA concat buffer aliases the first one).

#include <hip/hip_runtime.h>
#include <hip/hip_bf16.h>
#include <math.h>

#define NN    4096
#define CS    256
#define CZv   128
#define CHID  32
#define NH    8
#define NPQ   8
#define NPV   12
#define ESP   (NN * 32)
#define ESEQ  (NN * 4)

typedef __attribute__((ext_vector_type(16))) __bf16 bf16x16;
typedef __attribute__((ext_vector_type(8)))  float  f32x8;

union BFrag { bf16x16 v; __bf16 e[16]; float4 q[2]; };
union FAcc  { f32x8 v;   float  e[8]; };

// Load 8 contiguous f32 and convert to bf16 fragment elements.
__device__ __forceinline__ void loadA8(const float* __restrict__ p, __bf16* dst) {
  float4 u = *reinterpret_cast<const float4*>(p);
  float4 v = *reinterpret_cast<const float4*>(p + 4);
  dst[0] = (__bf16)u.x; dst[1] = (__bf16)u.y; dst[2] = (__bf16)u.z; dst[3] = (__bf16)u.w;
  dst[4] = (__bf16)v.x; dst[5] = (__bf16)v.y; dst[6] = (__bf16)v.z; dst[7] = (__bf16)v.w;
}

// ------------------------------------------------------------ weight pack
// Wp[(k/32)*(N*32) + n*32 + (k%32)] = bf16(W[k*N + n]); zero-pad k >= Ksrc.
// Lane fragment (group g) then reads 8 bf16 at n*32 + g*8 and 8 more at n*32 + g*8 + 16.
__global__ void k_packW(const float* __restrict__ W, __bf16* __restrict__ Wp,
                        int Ksrc, int Kpad, int N) {
  long i = (long)blockIdx.x * blockDim.x + threadIdx.x;
  if (i >= (long)Kpad * N) return;
  int k = (int)(i / N), n = (int)(i % N);
  float v = (k < Ksrc) ? W[(size_t)k * N + n] : 0.f;
  Wp[(size_t)(k >> 5) * (N * 32) + n * 32 + (k & 31)] = (__bf16)v;
}

// ---------------------------------------------------------------- WMMA GEMM
// C[M,N] = act(A[M,K] @ W[K,N] + bias). Row-major A (f32), packed bf16 W.
// K % 32 == 0, M % 64 == 0, N % 16 == 0. 128 threads = 4 waves; each wave
// computes a 64x16 tile of C (4 WMMA accumulators sharing one B fragment).
__global__ __launch_bounds__(128) void k_gemm(
    const float* __restrict__ A, const __bf16* __restrict__ Wp,
    const float* __restrict__ bias, float* __restrict__ C,
    int M, int K, int N, int act)
{
  const int lane = threadIdx.x & 31;
  const int wid  = threadIdx.x >> 5;
  const int tn   = blockIdx.x;
  const int row0 = (blockIdx.y * 4 + wid) * 64;          // wave supertile
  if (row0 >= M) return;
  const int ncol = tn * 16 + (lane & 15);
  const int grp  = lane >> 4;
  const float* Ar[4];
#pragma unroll
  for (int t = 0; t < 4; ++t)
    Ar[t] = A + (size_t)(row0 + t * 16 + (lane & 15)) * K;
  FAcc acc[4];
#pragma unroll
  for (int t = 0; t < 4; ++t)
#pragma unroll
    for (int r = 0; r < 8; ++r) acc[t].e[r] = 0.f;
  const int nslice = N * 32;
  for (int k0 = 0; k0 < K; k0 += 32) {
    BFrag b;
    const __bf16* wp = Wp + (size_t)(k0 >> 5) * nslice + ncol * 32 + grp * 8;
    b.q[0] = *reinterpret_cast<const float4*>(wp);
    b.q[1] = *reinterpret_cast<const float4*>(wp + 16);
#pragma unroll
    for (int t = 0; t < 4; ++t) {
      BFrag a;
      loadA8(Ar[t] + k0 + grp * 8,      &a.e[0]);
      loadA8(Ar[t] + k0 + grp * 8 + 16, &a.e[8]);
      acc[t].v = __builtin_amdgcn_wmma_f32_16x16x32_bf16(
          false, a.v, false, b.v, (short)0, acc[t].v, false, false);
    }
  }
  const float bv = bias ? bias[ncol] : 0.f;
#pragma unroll
  for (int t = 0; t < 4; ++t)
#pragma unroll
    for (int r = 0; r < 8; ++r) {
      int mm = row0 + t * 16 + r + 8 * grp;
      float v = acc[t].e[r] + bv;
      if (act) v = fmaxf(v, 0.f);
      C[(size_t)mm * N + ncol] = v;
    }
}

// Edge-transition layer-1 GEMM with fused gather of A = [z | h[src] | h[dst]]
// (K = 256 = 128 + 64 + 64, N = 256, ReLU). Segment boundaries are 32-aligned,
// so each 32-wide K slice picks one segment uniformly per step.
__global__ __launch_bounds__(128) void k_gemm_etcat(
    const float* __restrict__ Z, const float* __restrict__ HN,
    const int* __restrict__ srcI, const int* __restrict__ dstI,
    const __bf16* __restrict__ Wp, const float* __restrict__ bias,
    float* __restrict__ C, int M)
{
  const int N = 256;
  const int lane = threadIdx.x & 31;
  const int wid  = threadIdx.x >> 5;
  const int tn   = blockIdx.x;
  const int row0 = (blockIdx.y * 4 + wid) * 64;
  if (row0 >= M) return;
  const int ncol = tn * 16 + (lane & 15);
  const int grp  = lane >> 4;
  const float* pZ[4]; const float* pS[4]; const float* pD[4];
#pragma unroll
  for (int t = 0; t < 4; ++t) {
    int row = row0 + t * 16 + (lane & 15);
    pZ[t] = Z  + (size_t)row * 128;
    pS[t] = HN + (size_t)srcI[row] * 64;
    pD[t] = HN + (size_t)dstI[row] * 64;
  }
  FAcc acc[4];
#pragma unroll
  for (int t = 0; t < 4; ++t)
#pragma unroll
    for (int r = 0; r < 8; ++r) acc[t].e[r] = 0.f;
  for (int k0 = 0; k0 < 256; k0 += 32) {
    BFrag b;
    const __bf16* wp = Wp + (size_t)(k0 >> 5) * (N * 32) + ncol * 32 + grp * 8;
    b.q[0] = *reinterpret_cast<const float4*>(wp);
    b.q[1] = *reinterpret_cast<const float4*>(wp + 16);
    const int seg  = (k0 < 128) ? 0 : (k0 < 192) ? 1 : 2;      // uniform
    const int koff = k0 - ((seg == 0) ? 0 : (seg == 1) ? 128 : 192);
#pragma unroll
    for (int t = 0; t < 4; ++t) {
      const float* base = (seg == 0) ? pZ[t] : (seg == 1) ? pS[t] : pD[t];
      BFrag a;
      loadA8(base + koff + grp * 8,      &a.e[0]);
      loadA8(base + koff + grp * 8 + 16, &a.e[8]);
      acc[t].v = __builtin_amdgcn_wmma_f32_16x16x32_bf16(
          false, a.v, false, b.v, (short)0, acc[t].v, false, false);
    }
  }
  const float bv = bias[ncol];
#pragma unroll
  for (int t = 0; t < 4; ++t)
#pragma unroll
    for (int r = 0; r < 8; ++r) {
      int mm = row0 + t * 16 + r + 8 * grp;
      C[(size_t)mm * N + ncol] = fmaxf(acc[t].e[r] + bv, 0.f);
    }
}

// ---------------------------------------------------------------- LayerNorm
// out = LN( res? (res + x*rowscale[row]) : x ); one 32-lane wave per row.
__global__ __launch_bounds__(256) void k_ln(
    const float* __restrict__ x, const float* __restrict__ res,
    const float* __restrict__ rowscale, const float* __restrict__ g,
    const float* __restrict__ b, float* __restrict__ out, int rows, int width)
{
  const int row  = blockIdx.x * (blockDim.x >> 5) + (threadIdx.x >> 5);
  const int lane = threadIdx.x & 31;
  if (row >= rows) return;
  const int per = width >> 5;              // 4 (width 128) or 8 (width 256)
  float vals[8];
  float sum = 0.f;
  const float rs = rowscale ? rowscale[row] : 1.f;
  for (int i = 0; i < per; ++i) {
    int c = lane + i * 32;
    float v = x[(size_t)row * width + c] * rs;
    if (res) v += res[(size_t)row * width + c];
    vals[i] = v;
    sum += v;
  }
  for (int off = 16; off; off >>= 1) sum += __shfl_xor(sum, off, 32);
  const float mean = sum / (float)width;
  float var = 0.f;
  for (int i = 0; i < per; ++i) { float d = vals[i] - mean; var += d * d; }
  for (int off = 16; off; off >>= 1) var += __shfl_xor(var, off, 32);
  const float inv = rsqrtf(var / (float)width + 1e-5f);
  for (int i = 0; i < per; ++i) {
    int c = lane + i * 32;
    out[(size_t)row * width + c] = (vals[i] - mean) * inv * g[c] + b[c];
  }
}

// ---------------------------------------------------------------- utilities
__global__ void k_fill(float* __restrict__ p, float v, long n) {
  long i = (long)blockIdx.x * blockDim.x + threadIdx.x;
  if (i < n) p[i] = v;
}

__global__ void k_rowscale(float* __restrict__ p, const float* __restrict__ s,
                           int rows, int width) {
  long i = (long)blockIdx.x * blockDim.x + threadIdx.x;
  if (i >= (long)rows * width) return;
  p[i] *= s[i / width];
}

__global__ void k_center(const float* __restrict__ rig, float* __restrict__ center, int n) {
  __shared__ float sb[768];
  float sx = 0, sy = 0, sz = 0;
  for (int i = threadIdx.x; i < n; i += 256) {
    sx += rig[(size_t)i * 7 + 4];
    sy += rig[(size_t)i * 7 + 5];
    sz += rig[(size_t)i * 7 + 6];
  }
  sb[threadIdx.x] = sx; sb[256 + threadIdx.x] = sy; sb[512 + threadIdx.x] = sz;
  __syncthreads();
  for (int s = 128; s > 0; s >>= 1) {
    if ((int)threadIdx.x < s) {
      sb[threadIdx.x]       += sb[threadIdx.x + s];
      sb[256 + threadIdx.x] += sb[256 + threadIdx.x + s];
      sb[512 + threadIdx.x] += sb[512 + threadIdx.x + s];
    }
    __syncthreads();
  }
  if (threadIdx.x == 0) {
    center[0] = sb[0] / n; center[1] = sb[256] / n; center[2] = sb[512] / n;
  }
}

__global__ void k_prep(const float* __restrict__ rig, const float* __restrict__ center,
                       float* __restrict__ R, float* __restrict__ X,
                       float* __restrict__ t01, int n) {
  int i = blockIdx.x * blockDim.x + threadIdx.x;
  if (i >= n) return;
  float qw = rig[(size_t)i*7+0], qx = rig[(size_t)i*7+1];
  float qy = rig[(size_t)i*7+2], qz = rig[(size_t)i*7+3];
  float inv = rsqrtf(qw*qw + qx*qx + qy*qy + qz*qz + 1e-12f);
  qw *= inv; qx *= inv; qy *= inv; qz *= inv;
  float* Rn = R + (size_t)i * 9;
  Rn[0] = 1.f - 2.f*(qy*qy + qz*qz); Rn[1] = 2.f*(qx*qy - qw*qz); Rn[2] = 2.f*(qx*qz + qw*qy);
  Rn[3] = 2.f*(qx*qy + qw*qz); Rn[4] = 1.f - 2.f*(qx*qx + qz*qz); Rn[5] = 2.f*(qy*qz - qw*qx);
  Rn[6] = 2.f*(qx*qz - qw*qy); Rn[7] = 2.f*(qy*qz + qw*qx); Rn[8] = 1.f - 2.f*(qx*qx + qy*qy);
  for (int d = 0; d < 3; ++d) {
    float v = rig[(size_t)i*7 + 4 + d] - center[d];
    X[(size_t)i*3 + d]   = v;
    t01[(size_t)i*3 + d] = v * 0.1f;
  }
}

__global__ void k_edgefeat(const float* __restrict__ X, const int* __restrict__ residx,
                           const int* __restrict__ srcI, const int* __restrict__ dstI,
                           float* __restrict__ z, int E) {
  long idx = (long)blockIdx.x * blockDim.x + threadIdx.x;
  if (idx >= (long)E * 128) return;
  int e = (int)(idx >> 7), c = (int)(idx & 127);
  int sN = srcI[e], dN = dstI[e];
  float dx = X[(size_t)dN*3+0] - X[(size_t)sN*3+0];
  float dy = X[(size_t)dN*3+1] - X[(size_t)sN*3+1];
  float dz = X[(size_t)dN*3+2] - X[(size_t)sN*3+2];
  float dist = sqrtf(dx*dx + dy*dy + dz*dz + 1e-8f);
  float v;
  if (c < 64) {
    float mu = 20.f * (float)c / 63.f;
    float r = (dist - mu) / (20.f / 64.f);
    v = expf(-r * r);
  } else {
    int j = c - 64;
    int i = (j < 32) ? j : j - 32;
    float freq = expf(-(2.f * i) * (9.2103403719761836f / 64.f));
    float ang = (float)(residx[dN] - residx[sN]) * freq;
    v = (j < 32) ? cosf(ang) : sinf(ang);
  }
  z[idx] = v;
}

// node_in width 352: [res_pos(256) | tfeat(64) | nm(1) | zeros(31)]
__global__ void k_nodein(const int* __restrict__ residx, const float* __restrict__ ts,
                         const float* __restrict__ nm, float* __restrict__ out, int n) {
  long idx = (long)blockIdx.x * blockDim.x + threadIdx.x;
  if (idx >= (long)n * 352) return;
  int node = (int)(idx / 352), c = (int)(idx % 352);
  float v;
  if (c < 256) {
    int i = (c < 128) ? c : c - 128;
    float freq = expf(-(2.f * i) * (9.2103403719761836f / 256.f));
    float ang = (float)residx[node] * freq;
    v = (c < 128) ? cosf(ang) : sinf(ang);
  } else if (c < 320) {
    int j = c - 256;
    int i = (j < 32) ? j : j - 32;
    float freq = expf(-(2.f * i) * (9.2103403719761836f / 64.f));
    float ang = ts[node] * 1000.f * freq;
    v = (j < 32) ? cosf(ang) : sinf(ang);
  } else if (c == 320) {
    v = nm[node];
  } else {
    v = 0.f;
  }
  out[idx] = v;
}

// in-place R*p + t on packed (count = n*NH*P) points
__global__ void k_to_global(float* __restrict__ pts, const float* __restrict__ R,
                            const float* __restrict__ t, int n, int P) {
  long idx = (long)blockIdx.x * blockDim.x + threadIdx.x;
  long total = (long)n * NH * P;
  if (idx >= total) return;
  int node = (int)(idx / (NH * P));
  float* p = pts + idx * 3;
  float x = p[0], y = p[1], z = p[2];
  const float* Rn = R + (size_t)node * 9;
  const float* tn = t + (size_t)node * 3;
  p[0] = Rn[0]*x + Rn[1]*y + Rn[2]*z + tn[0];
  p[1] = Rn[3]*x + Rn[4]*y + Rn[5]*z + tn[1];
  p[2] = Rn[6]*x + Rn[7]*y + Rn[8]*z + tn[2];
}

__global__ void k_bz(const float* __restrict__ z, const float* __restrict__ W,
                     const float* __restrict__ b, float* __restrict__ out, int E) {
  long idx = (long)blockIdx.x * blockDim.x + threadIdx.x;
  if (idx >= (long)E * NH) return;
  int e = (int)(idx >> 3), h = (int)(idx & 7);
  float a = b[h];
  const float* zr = z + (size_t)e * CZv;
  for (int c = 0; c < CZv; ++c) a += zr[c] * W[c * NH + h];
  out[idx] = a;
}

__device__ __forceinline__ void atomicMaxF(float* addr, float val) {
  if (val >= 0.f) atomicMax((int*)addr, __float_as_int(val));
  else            atomicMin((unsigned int*)addr, __float_as_uint(val));
}

__global__ void k_att(const float* __restrict__ q, const float* __restrict__ kv,
                      const float* __restrict__ qp, const float* __restrict__ kvp,
                      const float* __restrict__ be, const float* __restrict__ mask,
                      const int* __restrict__ srcI, const int* __restrict__ dstI,
                      const float* __restrict__ gamma, float* __restrict__ att,
                      float* __restrict__ amax, int E) {
  long idx = (long)blockIdx.x * blockDim.x + threadIdx.x;
  if (idx >= (long)E * NH) return;
  int e = (int)(idx >> 3), h = (int)(idx & 7);
  int sN = srcI[e], dN = dstI[e];
  const float* qr = q  + (size_t)dN * 256 + h * 32;
  const float* kr = kv + (size_t)sN * 512 + h * 64;
  float a = 0.f;
  for (int c = 0; c < 32; ++c) a += qr[c] * kr[c];
  a *= 0.10206207261596577f;                       // sqrt(1/96)
  a += 0.5773502691896258f * be[idx];              // sqrt(1/3)
  const float* qpp = qp  + ((size_t)dN * NH + h) * (NPQ * 3);
  const float* kpp = kvp + ((size_t)sN * NH + h) * ((NPQ + NPV) * 3);
  float d2 = 0.f;
  for (int p = 0; p < NPQ * 3; ++p) { float d = qpp[p] - kpp[p]; d2 += d * d; }
  float gm = gamma[h];
  gm = (gm > 20.f) ? gm : log1pf(expf(gm));        // softplus
  a -= 0.5f * 0.09622504486493764f * gm * d2;      // sqrt(1/108)
  a += (mask[sN] - 1.f) * 1e9f;
  att[idx] = a;
  atomicMaxF(&amax[(size_t)dN * NH + h], a);
}

__global__ void k_exp(float* __restrict__ att, const float* __restrict__ amax,
                      const int* __restrict__ dstI, float* __restrict__ denom, int E) {
  long idx = (long)blockIdx.x * blockDim.x + threadIdx.x;
  if (idx >= (long)E * NH) return;
  int e = (int)(idx >> 3), h = (int)(idx & 7);
  int dN = dstI[e];
  float ex = expf(att[idx] - amax[(size_t)dN * NH + h]);
  att[idx] = ex;
  atomicAdd(&denom[(size_t)dN * NH + h], ex);
}

__global__ void k_accum(const float* __restrict__ att, const float* __restrict__ denom,
                        const float* __restrict__ kv, const float* __restrict__ kvp,
                        const float* __restrict__ z, const int* __restrict__ srcI,
                        const int* __restrict__ dstI, float* __restrict__ oA,
                        float* __restrict__ optA, float* __restrict__ opairA, int E) {
  long idx = (long)blockIdx.x * blockDim.x + threadIdx.x;
  if (idx >= (long)E * NH) return;
  int e = (int)(idx >> 3), h = (int)(idx & 7);
  int sN = srcI[e], dN = dstI[e];
  float w = att[idx] / (denom[(size_t)dN * NH + h] + 1e-9f);
  const float* vr = kv + (size_t)sN * 512 + h * 64 + 32;
  float* orow = oA + (size_t)dN * 256 + h * 32;
  for (int c = 0; c < 32; ++c) atomicAdd(&orow[c], w * vr[c]);
  const float* vp = kvp + ((size_t)sN * NH + h) * 60 + 24;   // points 8..19
  float* optr = optA + ((size_t)dN * NH + h) * 36;
  for (int j = 0; j < 36; ++j) atomicAdd(&optr[j], w * vp[j]);
  const float* zr = z + (size_t)e * CZv;
  float* opr = opairA + ((size_t)dN * NH + h) * CZv;
  for (int c = 0; c < CZv; ++c) atomicAdd(&opr[c], w * zr[c]);
}

// Assemble IPA concat row: [o(256) | opt_local(288) | ||opt||(96) | opair(1024)]
__global__ void k_cat(const float* __restrict__ oA, const float* __restrict__ optA,
                      const float* __restrict__ opairA, const float* __restrict__ R,
                      const float* __restrict__ t, float* __restrict__ cat, int n) {
  long idx = (long)blockIdx.x * blockDim.x + threadIdx.x;
  if (idx >= (long)n * NH) return;
  int node = (int)(idx >> 3), h = (int)(idx & 7);
  float* row = cat + (size_t)node * 1664;
  const float* orow = oA + (size_t)node * 256 + h * 32;
  for (int c = 0; c < 32; ++c) row[h * 32 + c] = orow[c];
  const float* Rn = R + (size_t)node * 9;
  const float* tn = t + (size_t)node * 3;
  const float* optr = optA + idx * 36;
  for (int p = 0; p < NPV; ++p) {
    float j0 = optr[p*3+0] - tn[0];
    float j1 = optr[p*3+1] - tn[1];
    float j2 = optr[p*3+2] - tn[2];
    float l0 = Rn[0]*j0 + Rn[3]*j1 + Rn[6]*j2;   // R^T
    float l1 = Rn[1]*j0 + Rn[4]*j1 + Rn[7]*j2;
    float l2 = Rn[2]*j0 + Rn[5]*j1 + Rn[8]*j2;
    int pb = 256 + (h * NPV + p) * 3;
    row[pb+0] = l0; row[pb+1] = l1; row[pb+2] = l2;
    row[256 + 288 + h * NPV + p] = sqrtf(l0*l0 + l1*l1 + l2*l2 + 1e-8f);
  }
  const float* opr = opairA + idx * CZv;
  for (int c = 0; c < CZv; ++c) row[640 + h * CZv + c] = opr[c];
}

__global__ void k_bb(const float* __restrict__ s, const float* __restrict__ nm,
                     const float* __restrict__ W, const float* __restrict__ b,
                     float* __restrict__ u, int n) {
  long idx = (long)blockIdx.x * blockDim.x + threadIdx.x;
  if (idx >= (long)n * 6) return;
  int node = (int)(idx / 6), j = (int)(idx % 6);
  float nmv = nm[node];
  float a = 0.f;
  const float* sr = s + (size_t)node * CS;
  for (int c = 0; c < CS; ++c) a += sr[c] * W[c * 6 + j];
  u[idx] = nmv * (nmv * a + b[j]);
}

__global__ void k_frame(const float* __restrict__ u, float* __restrict__ R,
                        float* __restrict__ t, int n) {
  int i = blockIdx.x * blockDim.x + threadIdx.x;
  if (i >= n) return;
  const float* un = u + (size_t)i * 6;
  float qw = 1.f, qx = un[0], qy = un[1], qz = un[2];
  float inv = rsqrtf(qw*qw + qx*qx + qy*qy + qz*qz + 1e-12f);
  qw *= inv; qx *= inv; qy *= inv; qz *= inv;
  float Q[9];
  Q[0] = 1.f - 2.f*(qy*qy + qz*qz); Q[1] = 2.f*(qx*qy - qw*qz); Q[2] = 2.f*(qx*qz + qw*qy);
  Q[3] = 2.f*(qx*qy + qw*qz); Q[4] = 1.f - 2.f*(qx*qx + qz*qz); Q[5] = 2.f*(qy*qz - qw*qx);
  Q[6] = 2.f*(qx*qz - qw*qy); Q[7] = 2.f*(qy*qz + qw*qx); Q[8] = 1.f - 2.f*(qx*qx + qy*qy);
  float* Rn = R + (size_t)i * 9;
  float* tn = t + (size_t)i * 3;
  float Ro[9];
  for (int k = 0; k < 9; ++k) Ro[k] = Rn[k];
  float ux = un[3], uy = un[4], uz = un[5];
  tn[0] += Ro[0]*ux + Ro[1]*uy + Ro[2]*uz;
  tn[1] += Ro[3]*ux + Ro[4]*uy + Ro[5]*uz;
  tn[2] += Ro[6]*ux + Ro[7]*uy + Ro[8]*uz;
  for (int r = 0; r < 3; ++r)
    for (int c = 0; c < 3; ++c)
      Rn[r*3+c] = Ro[r*3+0]*Q[0*3+c] + Ro[r*3+1]*Q[1*3+c] + Ro[r*3+2]*Q[2*3+c];
}

__global__ void k_psi(const float* __restrict__ h, const float* __restrict__ W,
                      const float* __restrict__ b, float* __restrict__ psi, int n) {
  int i = blockIdx.x * blockDim.x + threadIdx.x;
  if (i >= n) return;
  const float* hr = h + (size_t)i * CS;
  float p0 = b[0], p1 = b[1];
  for (int c = 0; c < CS; ++c) { p0 += hr[c] * W[c*2+0]; p1 += hr[c] * W[c*2+1]; }
  float inv = rsqrtf(p0*p0 + p1*p1 + 1e-8f);
  psi[(size_t)i*2+0] = p0 * inv;
  psi[(size_t)i*2+1] = p1 * inv;
}

__global__ void k_out(const float* __restrict__ s, const float* __restrict__ R,
                      const float* __restrict__ t, const float* __restrict__ center,
                      const float* __restrict__ psi, float* __restrict__ out, int n) {
  long idx = (long)blockIdx.x * blockDim.x + threadIdx.x;
  if (idx >= (long)n * 270) return;
  int node = (int)(idx / 270), c = (int)(idx % 270);
  float v;
  if (c < 256)      v = s[(size_t)node * CS + c];
  else if (c < 265) v = R[(size_t)node * 9 + (c - 256)];
  else if (c < 268) v = t[(size_t)node * 3 + (c - 265)] * 10.f + center[c - 265];
  else              v = psi[(size_t)node * 2 + (c - 268)];
  out[idx] = v;
}

// ------------------------------------------------------------------- host
struct LinP { const float *w, *b; };
struct LnPp { const float *g, *b; };
struct IpaP { LinP q, kv, qp, kvp, bl, out; const float* gamma; };
struct EtPp { LinP node, l1, l2, l3; LnPp ln; };
struct LayP { IpaP sp, seq; LnPp ln1, ln2, lnnt; LinP nt1, nt2, nt3, bb; EtPp et, set; };

struct Ws {
  float *center, *R, *X, *t, *s, *z, *zs, *nodein, *tmp1, *tmp2;
  float *q, *kv, *qp, *kvp, *be, *att, *amax, *denom;
  float *oA, *optA, *opairA, *upd, *hn, *u6, *psi, *bufA, *bufB, *cat;
  __bf16 *wpack;
};

static inline int cdiv(long a, long b) { return (int)((a + b - 1) / b); }

// GEMM with on-the-fly weight packing. Kpad = A width (mult of 32); weight
// rows beyond Ksrc are zero-padded during packing.
static void gemm2(hipStream_t st, Ws& w, const float* A, const float* W,
                  const float* b, float* C, int M, int Ksrc, int Kpad,
                  int N, int act) {
  k_packW<<<cdiv((long)Kpad * N, 256), 256, 0, st>>>(W, w.wpack, Ksrc, Kpad, N);
  dim3 g(N / 16, cdiv(M, 256));
  k_gemm<<<g, 128, 0, st>>>(A, w.wpack, b, C, M, Kpad, N, act);
}
static void gemm(hipStream_t st, Ws& w, const float* A, const float* W,
                 const float* b, float* C, int M, int K, int N, int act) {
  gemm2(st, w, A, W, b, C, M, K, K, N, act);
}
static void lnrun(hipStream_t st, const float* x, const float* res, const float* rs,
                  const LnPp& p, float* out, int rows, int width) {
  k_ln<<<cdiv(rows, 8), 256, 0, st>>>(x, res, rs, p.g, p.b, out, rows, width);
}
static void fill(hipStream_t st, float* p, float v, long n) {
  k_fill<<<cdiv(n, 256), 256, 0, st>>>(p, v, n);
}

static void run_ipa(hipStream_t st, const IpaP& P, const LnPp& lnp, float* s,
                    const float* zE, const int* srcI, const int* dstI, int E,
                    const float* R, const float* t, const float* mask, Ws& w) {
  gemm(st, w, s, P.q.w,   P.q.b,   w.q,   NN, CS, 256, 0);
  gemm(st, w, s, P.kv.w,  P.kv.b,  w.kv,  NN, CS, 512, 0);
  gemm(st, w, s, P.qp.w,  P.qp.b,  w.qp,  NN, CS, 192, 0);
  gemm(st, w, s, P.kvp.w, P.kvp.b, w.kvp, NN, CS, 480, 0);
  k_to_global<<<cdiv((long)NN*NH*NPQ, 256), 256, 0, st>>>(w.qp, R, t, NN, NPQ);
  k_to_global<<<cdiv((long)NN*NH*(NPQ+NPV), 256), 256, 0, st>>>(w.kvp, R, t, NN, NPQ+NPV);
  k_bz<<<cdiv((long)E*NH, 256), 256, 0, st>>>(zE, P.bl.w, P.bl.b, w.be, E);
  fill(st, w.amax, -1e30f, (long)NN * NH);
  fill(st, w.denom, 0.f, (long)NN * NH);
  fill(st, w.oA, 0.f, (long)NN * 256);
  fill(st, w.optA, 0.f, (long)NN * 288);
  fill(st, w.opairA, 0.f, (long)NN * 1024);
  long eh = (long)E * NH;
  k_att<<<cdiv(eh, 256), 256, 0, st>>>(w.q, w.kv, w.qp, w.kvp, w.be, mask, srcI,
                                       dstI, P.gamma, w.att, w.amax, E);
  k_exp<<<cdiv(eh, 256), 256, 0, st>>>(w.att, w.amax, dstI, w.denom, E);
  k_accum<<<cdiv(eh, 256), 256, 0, st>>>(w.att, w.denom, w.kv, w.kvp, zE, srcI,
                                         dstI, w.oA, w.optA, w.opairA, E);
  k_cat<<<cdiv((long)NN*NH, 256), 256, 0, st>>>(w.oA, w.optA, w.opairA, R, t, w.cat, NN);
  gemm(st, w, w.cat, P.out.w, P.out.b, w.upd, NN, 1664, 256, 0);
  lnrun(st, w.upd, s, mask, lnp, s, NN, 256);   // s = LN(s + upd*mask)
}

static void run_et(hipStream_t st, const EtPp& P, const float* s, float* zE,
                   const int* srcI, const int* dstI, int E, Ws& w) {
  gemm(st, w, s, P.node.w, P.node.b, w.hn, NN, CS, 64, 0);
  k_packW<<<cdiv((long)256 * 256, 256), 256, 0, st>>>(P.l1.w, w.wpack, 256, 256, 256);
  { dim3 g(256 / 16, cdiv(E, 256));
    k_gemm_etcat<<<g, 128, 0, st>>>(zE, w.hn, srcI, dstI, w.wpack, P.l1.b, w.bufA, E); }
  gemm(st, w, w.bufA, P.l2.w, P.l2.b, w.bufB, E, 256, 256, 1);
  gemm(st, w, w.bufB, P.l3.w, P.l3.b, w.bufA, E, 256, 128, 0);
  lnrun(st, w.bufA, zE, nullptr, P.ln, zE, E, 128);  // z = LN(z + e)
}

extern "C" void kernel_launch(void* const* d_in, const int* in_sizes, int n_in,
                              void* d_out, int out_size, void* d_ws, size_t ws_size,
                              hipStream_t stream) {
  (void)in_sizes; (void)n_in; (void)out_size; (void)ws_size;
  const float* ts     = (const float*)d_in[0];
  const int*   residx = (const int*)  d_in[1];
  const float* nm     = (const float*)d_in[2];
  const float* mask   = (const float*)d_in[3];
  const float* rig    = (const float*)d_in[4];
  const int*   eidx   = (const int*)  d_in[5];
  const int*   sidx   = (const int*)  d_in[6];
  const int* srcSP = eidx,  *dstSP = eidx + ESP;
  const int* srcSQ = sidx,  *dstSQ = sidx + ESEQ;

  // ---- parse params (depth-first insertion order) ----
  int pi = 7;
  auto nf  = [&]() { return (const float*)d_in[pi++]; };
  auto pl  = [&]() { LinP l; l.w = nf(); l.b = nf(); return l; };
  auto pln = [&]() { LnPp l; l.g = nf(); l.b = nf(); return l; };
  auto pip = [&]() { IpaP p; p.q = pl(); p.kv = pl(); p.qp = pl(); p.kvp = pl();
                     p.bl = pl(); p.gamma = nf(); p.out = pl(); return p; };
  auto pet = [&]() { EtPp e; e.node = pl(); e.l1 = pl(); e.l2 = pl(); e.l3 = pl();
                     e.ln = pln(); return e; };
  LinP en1 = pl(), en2 = pl(), en3 = pl();  LnPp enln = pln();
  LinP ee1 = pl(), ee2 = pl(), ee3 = pl();  LnPp eeln = pln();
  LayP L[2];
  for (int l = 0; l < 2; ++l) {
    L[l].sp  = pip(); L[l].seq = pip();
    L[l].ln1 = pln(); L[l].ln2 = pln();
    L[l].nt1 = pl();  L[l].nt2 = pl(); L[l].nt3 = pl();
    L[l].lnnt = pln();
    L[l].bb  = pl();
    L[l].et  = pet(); L[l].set = pet();
  }
  LinP ta1 = pl(), ta2 = pl(), ta3 = pl();

  // ---- workspace bump allocation ----
  char* base = (char*)d_ws; size_t off = 0;
  auto allocB = [&](size_t bytes) {
    void* p = base + off;
    off = (off + bytes + 255) & ~(size_t)255;
    return p;
  };
  auto alloc = [&](size_t nfloats) { return (float*)allocB(nfloats * 4); };
  Ws w;
  w.center = alloc(4);
  w.R      = alloc((size_t)NN * 9);
  w.X      = alloc((size_t)NN * 3);
  w.t      = alloc((size_t)NN * 3);
  w.s      = alloc((size_t)NN * CS);
  w.z      = alloc((size_t)ESP * CZv);
  w.zs     = alloc((size_t)ESEQ * CZv);
  w.nodein = alloc((size_t)NN * 352);
  w.tmp1   = alloc((size_t)NN * CS);
  w.tmp2   = alloc((size_t)NN * CS);
  w.q      = alloc((size_t)NN * 256);
  w.kv     = alloc((size_t)NN * 512);
  w.qp     = alloc((size_t)NN * 192);
  w.kvp    = alloc((size_t)NN * 480);
  w.be     = alloc((size_t)ESP * NH);
  w.att    = alloc((size_t)ESP * NH);
  w.amax   = alloc((size_t)NN * NH);
  w.denom  = alloc((size_t)NN * NH);
  w.oA     = alloc((size_t)NN * 256);
  w.optA   = alloc((size_t)NN * 288);
  w.opairA = alloc((size_t)NN * 1024);
  w.upd    = alloc((size_t)NN * 256);
  w.hn     = alloc((size_t)NN * 64);
  w.u6     = alloc((size_t)NN * 6);
  w.psi    = alloc((size_t)NN * 2);
  w.wpack  = (__bf16*)allocB((size_t)1664 * 256 * 2);
  w.bufA   = alloc((size_t)ESP * 256);
  w.bufB   = alloc((size_t)ESP * 256);
  w.cat    = w.bufA;   // alias: cat (NN x 1664) fits in bufA, disjoint lifetime

  hipStream_t st = stream;

  // ---- preprocessing ----
  k_center<<<1, 256, 0, st>>>(rig, w.center, NN);
  k_prep<<<cdiv(NN, 256), 256, 0, st>>>(rig, w.center, w.R, w.X, w.t, NN);
  k_edgefeat<<<cdiv((long)ESP * 128, 256), 256, 0, st>>>(w.X, residx, srcSP, dstSP, w.z, ESP);
  k_edgefeat<<<cdiv((long)ESEQ * 128, 256), 256, 0, st>>>(w.X, residx, srcSQ, dstSQ, w.zs, ESEQ);
  k_nodein<<<cdiv((long)NN * 352, 256), 256, 0, st>>>(residx, ts, nm, w.nodein, NN);

  // node encoder: s = LN(en3(relu(en2(relu(en1(node_in)))))) * mask
  gemm2(st, w, w.nodein, en1.w, en1.b, w.tmp1, NN, 321, 352, 256, 1);
  gemm(st, w, w.tmp1,   en2.w, en2.b, w.tmp2, NN, 256, 256, 1);
  gemm(st, w, w.tmp2,   en3.w, en3.b, w.tmp1, NN, 256, 256, 0);
  lnrun(st, w.tmp1, nullptr, nullptr, enln, w.s, NN, 256);
  k_rowscale<<<cdiv((long)NN * 256, 256), 256, 0, st>>>(w.s, mask, NN, 256);

  // edge encoder on z only
  gemm(st, w, w.z,    ee1.w, ee1.b, w.bufA, ESP, 128, 128, 1);
  gemm(st, w, w.bufA, ee2.w, ee2.b, w.bufB, ESP, 128, 128, 1);
  gemm(st, w, w.bufB, ee3.w, ee3.b, w.bufA, ESP, 128, 128, 0);
  lnrun(st, w.bufA, nullptr, nullptr, eeln, w.z, ESP, 128);

  // ---- layers ----
  for (int l = 0; l < 2; ++l) {
    run_ipa(st, L[l].sp,  L[l].ln1, w.s, w.z,  srcSP, dstSP, ESP,  w.R, w.t, mask, w);
    run_ipa(st, L[l].seq, L[l].ln2, w.s, w.zs, srcSQ, dstSQ, ESEQ, w.R, w.t, mask, w);
    // node transition
    gemm(st, w, w.s,    L[l].nt1.w, L[l].nt1.b, w.tmp1, NN, 256, 256, 1);
    gemm(st, w, w.tmp1, L[l].nt2.w, L[l].nt2.b, w.tmp2, NN, 256, 256, 1);
    gemm(st, w, w.tmp2, L[l].nt3.w, L[l].nt3.b, w.tmp1, NN, 256, 256, 0);
    lnrun(st, w.tmp1, w.s, nullptr, L[l].lnnt, w.s, NN, 256);
    k_rowscale<<<cdiv((long)NN * 256, 256), 256, 0, st>>>(w.s, mask, NN, 256);
    // backbone update
    k_bb<<<cdiv((long)NN * 6, 256), 256, 0, st>>>(w.s, nm, L[l].bb.w, L[l].bb.b, w.u6, NN);
    k_frame<<<cdiv(NN, 256), 256, 0, st>>>(w.u6, w.R, w.t, NN);
    // edge transitions
    run_et(st, L[l].et,  w.s, w.z,  srcSP, dstSP, ESP,  w);
    run_et(st, L[l].set, w.s, w.zs, srcSQ, dstSQ, ESEQ, w);
  }

  // ---- heads ----
  gemm(st, w, w.s,    ta1.w, ta1.b, w.tmp1, NN, 256, 256, 1);
  gemm(st, w, w.tmp1, ta2.w, ta2.b, w.tmp2, NN, 256, 256, 1);
  k_psi<<<cdiv(NN, 256), 256, 0, st>>>(w.tmp2, ta3.w, ta3.b, w.psi, NN);
  k_out<<<cdiv((long)NN * 270, 256), 256, 0, st>>>(w.s, w.R, w.t, w.center, w.psi,
                                                   (float*)d_out, NN);
}
